// LoraLayer_28527172780347
// MI455X (gfx1250) — matrix operand order repaired and checked by reading, not compiled, not run on hardware
//
#include <hip/hip_runtime.h>

// ---------------------------------------------------------------------------
// LoRA layer  out = X @ (W + SCALE * A@B)   folded into a single f16-WMMA GEMM
//   X  : [M=8192, K=2048]  fp32
//   W  : [K=2048, N=2048]  fp32   (W_orig reshaped [H, NH*HD], row-major)
//   A  : [K=2048, 4]       fp32
//   B  : [4, N=2048]       fp32
//   out: [M=8192, N=2048]  fp32
//
// Pass 1 folds the rank-4 LoRA update into an f16 transposed weight (8 MB ws).
// Pass 2: 128x256x32 block-tile WMMA GEMM, 8 wave32 waves, 64x64 per wave
// (16 v_wmma per wave per K-step), double-buffered LDS, ONE barrier per
// K-step.  Weight tiles are fetched by the Tensor Data Mover
// (tensor_load_to_lds, TENSORcnt); X tiles go through VGPRs for the f32->f16
// conversion.
// ---------------------------------------------------------------------------

typedef _Float16 v16h __attribute__((ext_vector_type(16)));
typedef _Float16 h8   __attribute__((ext_vector_type(8)));
typedef _Float16 h4   __attribute__((ext_vector_type(4)));
typedef float    v8f  __attribute__((ext_vector_type(8)));

typedef unsigned int u32x4 __attribute__((ext_vector_type(4)));
typedef int          i32x8 __attribute__((ext_vector_type(8)));
typedef int          i32x4 __attribute__((ext_vector_type(4)));

constexpr int Kdim = 2048;   // H (reduction dim)
constexpr int Ndim = 2048;   // NH*HD
constexpr int Mdim = 8192;   // B*S
constexpr float kScale = 4.0f / 4.0f;   // ALPHA / RANK

// ---------------------------------------------------------------------------
// Pass 1: W_eff^T[n][k] = f16( W[k][n] + SCALE * sum_r A[k][r] * B[r][n] )
// ---------------------------------------------------------------------------
__global__ __launch_bounds__(256) void lora_fold_weights(
    const float* __restrict__ W, const float* __restrict__ A,
    const float* __restrict__ Bk, _Float16* __restrict__ Wt)
{
    int idx = blockIdx.x * 256 + threadIdx.x;     // 0 .. 2048*2048-1
    int c = idx & (Kdim - 1);                     // reduction index (contiguous)
    int n = idx >> 11;                            // output column
    float acc = W[(size_t)c * Ndim + n];
#pragma unroll
    for (int r = 0; r < 4; ++r)
        acc += kScale * A[c * 4 + r] * Bk[r * Ndim + n];
    Wt[(size_t)n * Kdim + c] = (_Float16)acc;
}

// ---------------------------------------------------------------------------
// Pass 2: WMMA GEMM
// ---------------------------------------------------------------------------
constexpr int BM  = 128;
constexpr int BN  = 256;
constexpr int BK  = 32;
constexpr int LDH = 40;   // LDS row stride in halves (64 B data + 16 B pad)

// TDM: load one 32(K) x 256(N) f16 tile of W^T into LDS with 16B padding
// every 64B row.  D# per CDNA5 ISA ch.8 (groups 0/1; 2D tile -> groups 2/3 = 0).
__device__ static inline void tdm_load_w_tile(const _Float16* gsrc,
                                              unsigned int lds_off)
{
    unsigned long long ga = (unsigned long long)(uintptr_t)gsrc;

    u32x4 g0;
    g0[0] = 1u;                                   // count=1, no gather/restore
    g0[1] = lds_off;                              // lds_addr [63:32]
    g0[2] = (unsigned int)ga;                     // global_addr [95:64]
    g0[3] = (unsigned int)(ga >> 32) | (2u << 30);// global_addr[56:32] | type=2

    i32x8 g1;
    g1[0] = (1 << 16)      // data_size = 2 bytes
          | (1 << 20)      // pad_enable (LDS dest padding)
          | (3 << 22)      // pad_interval: every 16 DWORDs (64 B)
          | (3 << 25);     // pad_amount : 4 DWORDs (16 B)
    g1[1] = (int)(2048u << 16);   // tensor_dim0[15:0]=2048   @ [63:48]
    g1[2] = (int)(2048u << 16);   // tensor_dim1[15:0]=2048   @ [95:80]
    g1[3] = (int)(32u   << 16);   // tile_dim0 = 32 (K)       @ [127:112]
    g1[4] = 256;                  // tile_dim1 = 256 (N)      @ [143:128]
    g1[5] = 2048;                 // tensor_dim0_stride = 2048 elements
    g1[6] = 0;
    g1[7] = 0;

    i32x4 z4 = {0, 0, 0, 0};
    i32x8 z8 = {0, 0, 0, 0, 0, 0, 0, 0};
    // 6-arg form on this toolchain: (g0, g1, g2, g3, extra, cpol)
    __builtin_amdgcn_tensor_load_to_lds(g0, g1, z4, z4, z8, 0);
}

__global__ __launch_bounds__(256) void lora_gemm_wmma_f16(
    const float* __restrict__ X, const _Float16* __restrict__ Wt,
    float* __restrict__ Out)
{
    // double-buffered tiles: As 2*10240 B, Bs 2*20480 B  -> 61440 B LDS
    __shared__ __align__(16) _Float16 As[2][BM * LDH];   // X tile   [m][k]
    __shared__ __align__(16) _Float16 Bs[2][BN * LDH];   // W^T tile [n][k]

    const int tid  = threadIdx.x;
    const int lane = tid & 31;
    const int wave = tid >> 5;
    const int l15  = lane & 15;
    const int lhi  = lane >> 4;          // 0 or 1

    const int bm0 = blockIdx.y * BM;
    const int bn0 = blockIdx.x * BN;
    const int wm0 = (wave & 1) * 64;     // 2 waves along M, 64 rows each
    const int wn0 = (wave >> 1) * 64;    // 4 waves along N, 64 cols each

    // X global->reg staging assignments
    const int xr0 = tid >> 3;            // X row 0..31  (+p*32)
    const int xkc = (tid & 7) * 4;       // X k-offset (floats)

    float4 xr[4];

    auto load_x = [&](int k0) {
#pragma unroll
        for (int p = 0; p < 4; ++p)
            xr[p] = *(const float4*)&X[(size_t)(bm0 + xr0 + p * 32) * Kdim + k0 + xkc];
    };

    auto store_x = [&](int buf) {
#pragma unroll
        for (int p = 0; p < 4; ++p) {
            h4 h;
            h[0] = (_Float16)xr[p].x;  h[1] = (_Float16)xr[p].y;
            h[2] = (_Float16)xr[p].z;  h[3] = (_Float16)xr[p].w;
            *(h4*)&As[buf][(xr0 + p * 32) * LDH + xkc] = h;
        }
    };

    v8f acc[4][4];
#pragma unroll
    for (int mi = 0; mi < 4; ++mi)
#pragma unroll
        for (int ni = 0; ni < 4; ++ni)
            acc[mi][ni] = (v8f)0.0f;

    const int NK = Kdim / BK;            // 64 K-steps
    const _Float16* wtile = &Wt[(size_t)bn0 * Kdim];
    const unsigned int bs_off[2] = {
        (unsigned int)(uintptr_t)&Bs[0][0],
        (unsigned int)(uintptr_t)&Bs[1][0] };

    // prologue: TDM fetch of W tile 0, X tile 0 through VGPRs
    if (wave == 0) tdm_load_w_tile(wtile, bs_off[0]);
    load_x(0);
    store_x(0);

    for (int kt = 0; kt < NK; ++kt) {
        const int buf = kt & 1;

        if (kt + 1 < NK) load_x((kt + 1) * BK);       // overlap VMEM with WMMA

        if (wave == 0) __builtin_amdgcn_s_wait_tensorcnt(0);  // W tile[buf] done
        __syncthreads();   // (a) As/Bs[buf] visible to all
                           // (b) all waves done reading buf^1 -> TDM may refill

        if (kt + 1 < NK && wave == 0)
            tdm_load_w_tile(wtile + (size_t)(kt + 1) * BK, bs_off[buf ^ 1]);

        // --- fragment loads per ISA 7.12.2 layouts ---
        v16h a[4], b[4];
#pragma unroll
        for (int mi = 0; mi < 4; ++mi) {
            // A 16x32: lanes 0-15 -> K {0-7,16-23}; lanes 16-31 -> K {8-15,24-31}
            const _Float16* pa = &As[buf][(wm0 + mi * 16 + l15) * LDH + lhi * 8];
            h8 a0 = *(const h8*)pa;
            h8 a1 = *(const h8*)(pa + 16);
#pragma unroll
            for (int i = 0; i < 8; ++i) { a[mi][i] = a0[i]; a[mi][8 + i] = a1[i]; }
        }
#pragma unroll
        for (int ni = 0; ni < 4; ++ni) {
            // B 32x16: lanes 0-15 -> K {0-7,16-23}; lanes 16-31 -> K {8-15,24-31} (N = lane&15)
            const _Float16* pb = &Bs[buf][(wn0 + ni * 16 + l15) * LDH + lhi * 8];
            h8 b0 = *(const h8*)pb;
            h8 b1 = *(const h8*)(pb + 16);
#pragma unroll
            for (int i = 0; i < 8; ++i) { b[ni][i] = b0[i]; b[ni][8 + i] = b1[i]; }
        }

#pragma unroll
        for (int mi = 0; mi < 4; ++mi)
#pragma unroll
            for (int ni = 0; ni < 4; ++ni)
                acc[mi][ni] = __builtin_amdgcn_wmma_f32_16x16x32_f16(
                    /*neg_a=*/false, a[mi], /*neg_b=*/false, b[ni],
                    /*c_mod=*/(short)0, acc[mi][ni],
                    /*reuse_a=*/false, /*reuse_b=*/false);

        if (kt + 1 < NK) store_x(buf ^ 1);   // fill other X buffer
    }

    // --- epilogue: C/D 16x16 f32 layout: VGPR j -> M=j (lanes 0-15) / j+8 ---
#pragma unroll
    for (int mi = 0; mi < 4; ++mi)
#pragma unroll
        for (int ni = 0; ni < 4; ++ni) {
            const int gm = bm0 + wm0 + mi * 16 + lhi * 8;
            const int gn = bn0 + wn0 + ni * 16 + l15;
#pragma unroll
            for (int j = 0; j < 8; ++j)
                Out[(size_t)(gm + j) * Ndim + gn] = acc[mi][ni][j];
        }
}

// ---------------------------------------------------------------------------
extern "C" void kernel_launch(void* const* d_in, const int* in_sizes, int n_in,
                              void* d_out, int out_size, void* d_ws, size_t ws_size,
                              hipStream_t stream) {
    (void)in_sizes; (void)n_in; (void)out_size; (void)ws_size;
    const float* x  = (const float*)d_in[0];   // [4,2048,2048]
    const float* W  = (const float*)d_in[1];   // [2048,16,128]
    const float* A  = (const float*)d_in[2];   // [2048,4]
    const float* Bk = (const float*)d_in[3];   // [4,16,128]
    _Float16* Wt = (_Float16*)d_ws;            // 2048*2048 f16 = 8 MB scratch

    lora_fold_weights<<<(Kdim * Ndim) / 256, 256, 0, stream>>>(W, A, Bk, Wt);

    dim3 grid(Ndim / BN, Mdim / BM);           // 8 x 64 blocks
    lora_gemm_wmma_f16<<<grid, 256, 0, stream>>>(x, Wt, (float*)d_out);
}
